// TimeceptionBlockV3_16458314679016
// MI455X (gfx1250) — matrix operand order repaired and checked by reading, not compiled, not run on hardware
//
#include <hip/hip_runtime.h>
#include <hip/hip_bf16.h>

// ---------------------------------------------------------------------------
// Types for CDNA5 WMMA (wave32): D(f32 16x16) = A(bf16 16x32) x B(bf16 32x16) + C
// ---------------------------------------------------------------------------
typedef __attribute__((ext_vector_type(16))) __bf16        v16bf;
typedef __attribute__((ext_vector_type(8)))  float         v8f;
typedef __attribute__((ext_vector_type(4)))  unsigned int  u32x4;
typedef __attribute__((ext_vector_type(8)))  int           i32x8;
typedef __attribute__((ext_vector_type(4)))  int           i32x4;

struct U32x8 { u32x4 lo, hi; };

#define LDSK 40  // padded K-stride (bf16 elems): 80B rows keep 16B alignment, avoid bank aliasing

__device__ __forceinline__ unsigned short f32_to_bf16_rne(float f) {
    unsigned int u = __float_as_uint(f);
    u += 0x7FFFu + ((u >> 16) & 1u);
    return (unsigned short)(u >> 16);
}

__device__ __forceinline__ unsigned int pack2_bf16(float lo, float hi) {
    return (unsigned int)f32_to_bf16_rne(lo) | ((unsigned int)f32_to_bf16_rne(hi) << 16);
}

// A fragment: lane L holds row M = base + (L&15); K half = (L<16)?0:8 for V0-3, +16 for V4-7.
__device__ __forceinline__ v16bf frag_a(const unsigned short* lds, int mrow, int lane) {
    const int khalf = (lane & 16) ? 8 : 0;
    const unsigned short* p = lds + mrow * LDSK + khalf;
    U32x8 t;
    t.lo = *(const u32x4*)(p);        // K = khalf .. khalf+7
    t.hi = *(const u32x4*)(p + 16);   // K = khalf+16 .. khalf+23
    return __builtin_bit_cast(v16bf, t);
}

// B fragment: lane L holds col N = base + (L&15); K half = (L<16)?0:8 for V0-3, +16 for V4-7.
__device__ __forceinline__ v16bf frag_b(const unsigned short* lds, int ncol, int lane) {
    const int kbase = (lane & 16) ? 8 : 0;
    const unsigned short* p = lds + ncol * LDSK + kbase;
    U32x8 t;
    t.lo = *(const u32x4*)(p);
    t.hi = *(const u32x4*)(p + 16);
    return __builtin_bit_cast(v16bf, t);
}

// ---------------------------------------------------------------------------
// TDM: DMA one A-tile (128 rows x 32 bf16) from global [rows][ktot] into LDS,
// padding +16B after each 64B row => LDS row stride = 80B = LDSK bf16 elems.
// Descriptor packing per CDNA5 ISA (2D tensor:
// groups 2/3 zero). Issued by one wave; EXEC is ignored for tensor ops.
// This toolchain's builtin is the 6-arg flavor:
//   (uint32x4 g0, int32x8 g1, int32x4 g2, int32x4 g3, int32x8 extra, i32 cpol)
// ---------------------------------------------------------------------------
#if defined(__has_builtin)
#if __has_builtin(__builtin_amdgcn_tensor_load_to_lds) && __has_builtin(__builtin_amdgcn_s_wait_tensorcnt)
#define USE_TDM 1
#endif
#endif

__device__ __forceinline__ void tdm_load_tile_a(unsigned int lds_addr,
                                                const unsigned short* gtile,
                                                unsigned int ktot) {
#ifdef USE_TDM
    const unsigned long long ga = (unsigned long long)(size_t)gtile;
    u32x4 g0;
    g0[0] = 1u;                                                   // count=1, user mode
    g0[1] = lds_addr;                                             // lds_addr [63:32]
    g0[2] = (unsigned int)ga;                                     // global_addr [95:64]
    g0[3] = (unsigned int)((ga >> 32) & 0x01FFFFFFull) | (2u << 30); // addr[56:32] | type=2
    i32x8 g1;
    g1[0] = (1 << 16)        // data_size = 1 -> 2 bytes
          | (1 << 20)        // pad_enable
          | (3 << 22)        // pad_interval = 16 DWORDs (64B row)
          | (3 << 25);       // pad_amount   = 4 DWORDs (16B)
    g1[1] = (int)(ktot << 16);                              // tensor_dim0[15:0] @ [63:48]
    g1[2] = (int)((ktot >> 16) & 0xFFFFu) | (512 << 16);    // tensor_dim0 hi | tensor_dim1=512
    g1[3] = (int)(32u << 16);                               // tensor_dim1 hi=0 | tile_dim0=32
    g1[4] = 128;                                            // tile_dim1=128, tile_dim2=0
    g1[5] = (int)ktot;                                      // tensor_dim0_stride[31:0]
    g1[6] = 0;                                              // stride hi | dim1_stride lo
    g1[7] = 0;
    i32x4 z4 = {};
    i32x8 z8 = {};
    __builtin_amdgcn_tensor_load_to_lds(g0, g1, z4, z4, z8, 0);
#else
    (void)lds_addr; (void)gtile; (void)ktot;
#endif
}

__device__ __forceinline__ void tdm_wait() {
#ifdef USE_TDM
    __builtin_amdgcn_s_wait_tensorcnt(0);
#endif
}

#ifndef USE_TDM
// Fallback: cooperative copy of the A tile (t = 0..255).
#define MANUAL_A_TILE(lA_, A_, m0_, k0_, ktot_, t_)                                   \
    do {                                                                              \
        const int row_ = (t_) >> 1;                                                   \
        const int kh_  = ((t_) & 1) * 16;                                             \
        const unsigned short* src_ = (A_) + (size_t)((m0_) + row_) * (ktot_) + (k0_) + kh_; \
        unsigned short* dst_ = (lA_) + row_ * LDSK + kh_;                             \
        *(u32x4*)(dst_)     = *(const u32x4*)(src_);                                  \
        *(u32x4*)(dst_ + 8) = *(const u32x4*)(src_ + 8);                              \
    } while (0)
#endif

// ---------------------------------------------------------------------------
// Prep: A1 = bf16(conv1_w) [512x512];
//       Mw[o][g*512 + j*128 + c] = sum_k conv2_w[o][g*128+j*32+k] * conv_tcp_w[j][k][c]
//       bias_comb[o] = conv2_b[o] + sum_ic conv2_w[o][ic] * conv_tcp_b[ic & 127]
// ---------------------------------------------------------------------------
__global__ __launch_bounds__(256)
void prep_kernel(const float* __restrict__ conv1_w,
                 const float* __restrict__ conv_tcp_w,
                 const float* __restrict__ conv_tcp_b,
                 const float* __restrict__ conv2_w,
                 const float* __restrict__ conv2_b,
                 unsigned short* __restrict__ A1,
                 unsigned short* __restrict__ Mw,
                 float* __restrict__ bias_comb)
{
    const int i = blockIdx.x * blockDim.x + threadIdx.x;   // < 512*2048
    if (i >= 512 * 2048) return;
    const int o   = i >> 11;
    const int col = i & 2047;
    const int g   = col >> 9;          // group
    const int j   = (col >> 7) & 3;    // prefix-stack index
    const int c   = col & 127;         // channel within group
    float s = 0.0f;
    #pragma unroll 8
    for (int k = 0; k < 32; ++k)
        s += conv2_w[o * 512 + g * 128 + j * 32 + k] * conv_tcp_w[(j * 32 + k) * 128 + c];
    Mw[i] = f32_to_bf16_rne(s);
    if (col < 512)
        A1[o * 512 + col] = f32_to_bf16_rne(conv1_w[o * 512 + col]);
    if (col == 0) {
        float bs = conv2_b[o];
        for (int ic = 0; ic < 512; ++ic)
            bs += conv2_w[o * 512 + ic] * conv_tcp_b[ic & 127];
        bias_comb[o] = bs;
    }
}

// ---------------------------------------------------------------------------
// GEMM1: Y[img][o][hw] = conv1_w[o][c] * x[img][c][hw] + conv1_b[o]
// Block: 128(M) x 128(N), K-step 32; 8 waves, each 32x64 (2x4 WMMA tiles).
// A tile via TDM (wave 0); B tile: transpose+cvt staging, dword-packed.
// ---------------------------------------------------------------------------
__global__ __launch_bounds__(256)
void gemm1_kernel(const unsigned short* __restrict__ A,  // bf16 [512][512]
                  const float* __restrict__ X,           // f32 [64][512][784]
                  const float* __restrict__ bias,        // [512]
                  float* __restrict__ Y)                 // f32 [64][512][784]
{
    __shared__ alignas(16) unsigned short lA[128 * LDSK];
    __shared__ alignas(16) unsigned short lB[128 * LDSK];

    const int t    = threadIdx.x;
    const int lane = t & 31;
    const int wave = t >> 5;
    const int wm   = wave >> 1;   // 0..3
    const int wn   = wave & 1;    // 0..1

    const int n0  = blockIdx.x * 128;   // col tile within 784
    const int m0  = blockIdx.y * 128;   // row tile within 512
    const int img = blockIdx.z;
    const bool full = (n0 + 128 <= 784);

    const float* Xi = X + (size_t)img * 512 * 784;
    float*       Yi = Y + (size_t)img * 512 * 784;
    const unsigned int ldsA = (unsigned int)(size_t)lA;

    v8f acc[2][4];
    #pragma unroll
    for (int mi = 0; mi < 2; ++mi)
        #pragma unroll
        for (int ni = 0; ni < 4; ++ni) { v8f z{}; acc[mi][ni] = z; }

    for (int k0 = 0; k0 < 512; k0 += 32) {
#ifdef USE_TDM
        if (wave == 0)
            tdm_load_tile_a(ldsA, A + (size_t)m0 * 512 + k0, 512u);
#else
        MANUAL_A_TILE(lA, A, m0, k0, 512, t);
#endif
        // B tile: LDS[n][k] <- cvt(X[k][n]); pack k-pairs into dword stores.
        const int nl = t & 127;
        const int kh = (t >> 7) * 2;
        if (full) {
            #pragma unroll
            for (int kk = 0; kk < 32; kk += 4) {
                const int kl = kk + kh;
                const float v0 = Xi[(size_t)(k0 + kl) * 784 + n0 + nl];
                const float v1 = Xi[(size_t)(k0 + kl + 1) * 784 + n0 + nl];
                *(unsigned int*)&lB[nl * LDSK + kl] = pack2_bf16(v0, v1);
            }
        } else {
            const bool ok = (n0 + nl) < 784;
            #pragma unroll
            for (int kk = 0; kk < 32; kk += 4) {
                const int kl = kk + kh;
                const float v0 = ok ? Xi[(size_t)(k0 + kl) * 784 + n0 + nl] : 0.0f;
                const float v1 = ok ? Xi[(size_t)(k0 + kl + 1) * 784 + n0 + nl] : 0.0f;
                *(unsigned int*)&lB[nl * LDSK + kl] = pack2_bf16(v0, v1);
            }
        }
        if (wave == 0) tdm_wait();
        __syncthreads();

        v16bf bfr[4];
        #pragma unroll
        for (int ni = 0; ni < 4; ++ni)
            bfr[ni] = frag_b(lB, wn * 64 + ni * 16 + (lane & 15), lane);
        #pragma unroll
        for (int mi = 0; mi < 2; ++mi) {
            const v16bf afr = frag_a(lA, wm * 32 + mi * 16 + (lane & 15), lane);
            #pragma unroll
            for (int ni = 0; ni < 4; ++ni)
                acc[mi][ni] = __builtin_amdgcn_wmma_f32_16x16x32_bf16(
                    false, afr, false, bfr[ni], (short)0, acc[mi][ni], false, false);
        }
        __syncthreads();
    }

    // Epilogue: C/D layout -> lane holds N=lane&15, rows r (+8 for upper half-wave)
    #pragma unroll
    for (int mi = 0; mi < 2; ++mi) {
        #pragma unroll
        for (int ni = 0; ni < 4; ++ni) {
            const int col = n0 + wn * 64 + ni * 16 + (lane & 15);
            if (col < 784) {
                #pragma unroll
                for (int r = 0; r < 8; ++r) {
                    const int row = m0 + wm * 32 + mi * 16 + r + ((lane & 16) ? 8 : 0);
                    Yi[(size_t)row * 784 + col] = acc[mi][ni][r] + bias[row];
                }
            }
        }
    }
}

// ---------------------------------------------------------------------------
// Temporal: per (b,hw,c): depthwise convs over T=8 (k=1,3,5,7, zero pad) + prefix sums.
// Writes bf16 S[kidx][p], kidx = g*512 + j*128 + (c&127), p = (b*8+t)*784 + hw.
// ---------------------------------------------------------------------------
__global__ __launch_bounds__(256)
void temporal_kernel(const float* __restrict__ Y,   // [64][512][784]
                     const float* __restrict__ w1, const float* __restrict__ b1,
                     const float* __restrict__ w3, const float* __restrict__ b3,
                     const float* __restrict__ w5, const float* __restrict__ b5,
                     const float* __restrict__ w7, const float* __restrict__ b7,
                     unsigned short* __restrict__ S) // [2048][50176]
{
    const int i = blockIdx.x * blockDim.x + threadIdx.x;  // < 8*512*784
    if (i >= 8 * 512 * 784) return;
    const int hw = i % 784;
    const int c  = (i / 784) & 511;
    const int b  = i / (784 * 512);
    const int cl = c & 127;
    const int g  = c >> 7;

    float y[8];
    #pragma unroll
    for (int tt = 0; tt < 8; ++tt)
        y[tt] = Y[((size_t)(b * 8 + tt) * 512 + c) * 784 + hw];

    const float k1 = w1[cl];
    float t3[3], t5[5], t7[7];
    #pragma unroll
    for (int k = 0; k < 3; ++k) t3[k] = w3[cl * 3 + k];
    #pragma unroll
    for (int k = 0; k < 5; ++k) t5[k] = w5[cl * 5 + k];
    #pragma unroll
    for (int k = 0; k < 7; ++k) t7[k] = w7[cl * 7 + k];
    const float bb1 = b1[cl], bb3 = b3[cl], bb5 = b5[cl], bb7 = b7[cl];

    const size_t kbase = (size_t)(g * 512 + cl) * 50176;
    #pragma unroll
    for (int tt = 0; tt < 8; ++tt) {
        float o1 = k1 * y[tt] + bb1;
        float o2 = bb3, o3 = bb5, o4 = bb7;
        #pragma unroll
        for (int k = 0; k < 3; ++k) { const int s = tt + k - 1; if (s >= 0 && s < 8) o2 += t3[k] * y[s]; }
        #pragma unroll
        for (int k = 0; k < 5; ++k) { const int s = tt + k - 2; if (s >= 0 && s < 8) o3 += t5[k] * y[s]; }
        #pragma unroll
        for (int k = 0; k < 7; ++k) { const int s = tt + k - 3; if (s >= 0 && s < 8) o4 += t7[k] * y[s]; }
        const size_t p = (size_t)(b * 8 + tt) * 784 + hw;
        float s = o1;
        S[kbase + (size_t)(0 * 128) * 50176 + p] = f32_to_bf16_rne(s);
        s += o2;
        S[kbase + (size_t)(1 * 128) * 50176 + p] = f32_to_bf16_rne(s);
        s += o3;
        S[kbase + (size_t)(2 * 128) * 50176 + p] = f32_to_bf16_rne(s);
        s += o4;
        S[kbase + (size_t)(3 * 128) * 50176 + p] = f32_to_bf16_rne(s);
    }
}

// ---------------------------------------------------------------------------
// GEMM2: out[n][o][hw] = Mw[o][k2048] * S[k2048][p] + bias_comb[o] + x[n][o][hw]
// p = n*784 + hw, P = 50176 (= 392 * 128 tiles exactly). A tile via TDM.
// ---------------------------------------------------------------------------
__global__ __launch_bounds__(256)
void gemm2_kernel(const unsigned short* __restrict__ Mw,  // bf16 [512][2048]
                  const unsigned short* __restrict__ S,   // bf16 [2048][50176]
                  const float* __restrict__ bias_comb,    // [512]
                  const float* __restrict__ X,            // residual [64][512][784]
                  float* __restrict__ Out)                // [64][512][784]
{
    __shared__ alignas(16) unsigned short lA[128 * LDSK];
    __shared__ alignas(16) unsigned short lB[128 * LDSK];

    const int t    = threadIdx.x;
    const int lane = t & 31;
    const int wave = t >> 5;
    const int wm   = wave >> 1;
    const int wn   = wave & 1;

    const int p0 = blockIdx.x * 128;   // pixel tile
    const int m0 = blockIdx.y * 128;   // out-channel tile
    const unsigned int ldsA = (unsigned int)(size_t)lA;

    v8f acc[2][4];
    #pragma unroll
    for (int mi = 0; mi < 2; ++mi)
        #pragma unroll
        for (int ni = 0; ni < 4; ++ni) { v8f z{}; acc[mi][ni] = z; }

    for (int k0 = 0; k0 < 2048; k0 += 32) {
#ifdef USE_TDM
        if (wave == 0)
            tdm_load_tile_a(ldsA, Mw + (size_t)m0 * 2048 + k0, 2048u);
#else
        MANUAL_A_TILE(lA, Mw, m0, k0, 2048, t);
#endif
        // B tile: LDS[n][k] <- S[k][p0+n]; pack k-pairs into dword stores.
        {
            const int nl = t & 127;
            const int kh = (t >> 7) * 2;
            #pragma unroll
            for (int kk = 0; kk < 32; kk += 4) {
                const int kl = kk + kh;
                const unsigned int s0 = S[(size_t)(k0 + kl) * 50176 + p0 + nl];
                const unsigned int s1 = S[(size_t)(k0 + kl + 1) * 50176 + p0 + nl];
                *(unsigned int*)&lB[nl * LDSK + kl] = s0 | (s1 << 16);
            }
        }
        if (wave == 0) tdm_wait();
        __syncthreads();

        v16bf bfr[4];
        #pragma unroll
        for (int ni = 0; ni < 4; ++ni)
            bfr[ni] = frag_b(lB, wn * 64 + ni * 16 + (lane & 15), lane);
        #pragma unroll
        for (int mi = 0; mi < 2; ++mi) {
            const v16bf afr = frag_a(lA, wm * 32 + mi * 16 + (lane & 15), lane);
            #pragma unroll
            for (int ni = 0; ni < 4; ++ni)
                acc[mi][ni] = __builtin_amdgcn_wmma_f32_16x16x32_bf16(
                    false, afr, false, bfr[ni], (short)0, acc[mi][ni], false, false);
        }
        __syncthreads();
    }

    #pragma unroll
    for (int mi = 0; mi < 2; ++mi) {
        #pragma unroll
        for (int ni = 0; ni < 4; ++ni) {
            const int p     = p0 + wn * 64 + ni * 16 + (lane & 15);
            const int n_img = p / 784;
            const int hw    = p - n_img * 784;
            #pragma unroll
            for (int r = 0; r < 8; ++r) {
                const int row = m0 + wm * 32 + mi * 16 + r + ((lane & 16) ? 8 : 0);
                const size_t idx = ((size_t)n_img * 512 + row) * 784 + hw;
                Out[idx] = acc[mi][ni][r] + bias_comb[row] + X[idx];
            }
        }
    }
}

// ---------------------------------------------------------------------------
// Host-side launcher. Workspace layout (bytes):
//   A1 bf16   [512*512]      @ 0            (524,288)
//   Mw bf16   [512*2048]     @ 524288       (2,097,152)
//   bias f32  [512]          @ 2621440      (2,048)
//   Y  f32    [64*512*784]   @ 2623488      (102,760,448)
//   S  bf16   [2048*50176]   @ 105383936    (205,520,896)
//   total ~ 311 MB
// ---------------------------------------------------------------------------
extern "C" void kernel_launch(void* const* d_in, const int* in_sizes, int n_in,
                              void* d_out, int out_size, void* d_ws, size_t ws_size,
                              hipStream_t stream) {
    const float* x          = (const float*)d_in[0];
    const float* conv1_w    = (const float*)d_in[1];
    const float* conv1_b    = (const float*)d_in[2];
    const float* tcp_w1     = (const float*)d_in[3];
    const float* tcp_b1     = (const float*)d_in[4];
    const float* tcp_w3     = (const float*)d_in[5];
    const float* tcp_b3     = (const float*)d_in[6];
    const float* tcp_w5     = (const float*)d_in[7];
    const float* tcp_b5     = (const float*)d_in[8];
    const float* tcp_w7     = (const float*)d_in[9];
    const float* tcp_b7     = (const float*)d_in[10];
    const float* conv_tcp_w = (const float*)d_in[11];
    const float* conv_tcp_b = (const float*)d_in[12];
    const float* conv2_w    = (const float*)d_in[13];
    const float* conv2_b    = (const float*)d_in[14];
    float* out = (float*)d_out;

    char* ws = (char*)d_ws;
    unsigned short* A1    = (unsigned short*)(ws);
    unsigned short* Mw    = (unsigned short*)(ws + 524288);
    float*          biasc = (float*)         (ws + 2621440);
    float*          Y     = (float*)         (ws + 2623488);
    unsigned short* S     = (unsigned short*)(ws + 105383936);

    prep_kernel<<<(512 * 2048) / 256, 256, 0, stream>>>(
        conv1_w, conv_tcp_w, conv_tcp_b, conv2_w, conv2_b, A1, Mw, biasc);

    gemm1_kernel<<<dim3(7, 4, 64), 256, 0, stream>>>(A1, x, conv1_b, Y);

    temporal_kernel<<<(8 * 512 * 784) / 256, 256, 0, stream>>>(
        Y, tcp_w1, tcp_b1, tcp_w3, tcp_b3, tcp_w5, tcp_b5, tcp_w7, tcp_b7, S);

    gemm2_kernel<<<dim3(392, 4, 1), 256, 0, stream>>>(Mw, S, biasc, x, out);
}
